// LinearProbe_PerScene_27711128994544
// MI455X (gfx1250) — compile-verified, matrix operand order checked by reading, not run on hardware
//
#include <hip/hip_runtime.h>
#include <hip/hip_bf16.h>

// ---------------------------------------------------------------------------
// MI455X (gfx1250) fused per-scene linear probe:
//   sort-by-scene (counting sort) -> gather+fp16 convert -> scene GEMM (relu)
//   -> dense GEMM -> fp32 out.  Matmuls via v_wmma_f32_16x16x32_f16, K panels
//   staged with global_load_async_to_lds_b128 (ASYNCcnt) into double-buffered
//   LDS.
// ---------------------------------------------------------------------------

typedef _Float16 half_t;
typedef __attribute__((ext_vector_type(16))) _Float16 v16h;
typedef __attribute__((ext_vector_type(8)))  float    v8f;
typedef __attribute__((ext_vector_type(4)))  int      v4i;

#define N_ROWS 65536
#define DDIM   2048
#define HDIM   4096
#define NSCENE 4

#define TM 128                 // block tile rows
#define TN 128                 // block tile cols
#define TK 32                  // K panel (one f16 WMMA K)
#define KSTEPS (DDIM / TK)     // 64
#define LDSW 40                // padded row stride in halfs (80B: 16B aligned, conflict-free)

#define AS1 __attribute__((address_space(1)))
#define AS3 __attribute__((address_space(3)))

// ---------------------------------------------------------------------------
// CDNA5 async copy helpers: global -> LDS without VGPR staging (ASYNCcnt).
// ---------------------------------------------------------------------------
__device__ __forceinline__ void async_copy_b128(const half_t* g, half_t* l) {
  AS1 v4i* gp = (AS1 v4i*)(uintptr_t)g;
  AS3 v4i* lp = (AS3 v4i*)(unsigned int)(uintptr_t)l;
#if __has_builtin(__builtin_amdgcn_global_load_async_to_lds_b128)
  __builtin_amdgcn_global_load_async_to_lds_b128(gp, lp, 0, 0);
#else
  asm volatile("global_load_async_to_lds_b128 %0, %1, off"
               :: "v"(lp), "v"(gp) : "memory");
#endif
}

__device__ __forceinline__ void wait_async0() {
#if __has_builtin(__builtin_amdgcn_s_wait_asynccnt)
  __builtin_amdgcn_s_wait_asynccnt(0);
#else
  asm volatile("s_wait_asynccnt 0x0" ::: "memory");
#endif
}

// ---------------------------------------------------------------------------
// Stable counting sort by scene (single workgroup). Produces:
//   sorted_idx[N]  : original row for each sorted position
//   meta[0..4]     : group bounds g[0..4] (g[0]=0, g[4]=N)
//   meta[5..8]     : first M-tile index per scene (tiles of TM rows, per scene)
//   meta[9]        : total M-tiles
// ---------------------------------------------------------------------------
__global__ __launch_bounds__(256) void sort_scenes(const int* __restrict__ scene,
                                                   int* __restrict__ sorted_idx,
                                                   int* __restrict__ meta) {
  __shared__ int cnt[256][NSCENE];
  __shared__ int start[256][NSCENE];
  const int t = threadIdx.x;
  const int chunk = N_ROWS / 256;
  const int base = t * chunk;
  int c[NSCENE] = {0, 0, 0, 0};
  for (int i = 0; i < chunk; ++i) {
    int s = (scene[base + i] - 1) & 3;
    c[s]++;
  }
  for (int s = 0; s < NSCENE; ++s) cnt[t][s] = c[s];
  __syncthreads();
  if (t == 0) {
    int tot[NSCENE] = {0, 0, 0, 0};
    for (int s = 0; s < NSCENE; ++s)
      for (int b = 0; b < 256; ++b) tot[s] += cnt[b][s];
    int g[NSCENE + 1];
    g[0] = 0;
    for (int s = 0; s < NSCENE; ++s) g[s + 1] = g[s] + tot[s];
    int run[NSCENE];
    for (int s = 0; s < NSCENE; ++s) run[s] = g[s];
    for (int b = 0; b < 256; ++b)
      for (int s = 0; s < NSCENE; ++s) {
        start[b][s] = run[s];
        run[s] += cnt[b][s];
      }
    int ts = 0;
    for (int s = 0; s < NSCENE; ++s) {
      meta[s] = g[s];
      meta[5 + s] = ts;
      ts += (tot[s] + TM - 1) / TM;
    }
    meta[4] = g[NSCENE];
    meta[9] = ts;
  }
  __syncthreads();
  int run[NSCENE];
  for (int s = 0; s < NSCENE; ++s) run[s] = start[t][s];
  for (int i = 0; i < chunk; ++i) {
    int row = base + i;
    int s = (scene[row] - 1) & 3;
    sorted_idx[run[s]++] = row;
  }
}

// Gather rows into sorted order while converting fp32 -> fp16.
__global__ __launch_bounds__(256) void gather_convert_x(const float* __restrict__ x,
                                                        const int* __restrict__ sorted_idx,
                                                        half_t* __restrict__ dst) {
  const int p = blockIdx.x;
  const int srow = sorted_idx[p];
  const float* src = x + (size_t)srow * DDIM;
  half_t* d = dst + (size_t)p * DDIM;
  for (int k = threadIdx.x * 8; k < DDIM; k += blockDim.x * 8) {
    float4 a = *(const float4*)(src + k);
    float4 b = *(const float4*)(src + k + 4);
    union { half_t h[8]; uint4 q; } u;
    u.h[0] = (half_t)a.x; u.h[1] = (half_t)a.y; u.h[2] = (half_t)a.z; u.h[3] = (half_t)a.w;
    u.h[4] = (half_t)b.x; u.h[5] = (half_t)b.y; u.h[6] = (half_t)b.z; u.h[7] = (half_t)b.w;
    *(uint4*)(d + k) = u.q;
  }
}

// Flat fp32 -> fp16 conversion (weights; they then live in the 192MB L2).
__global__ __launch_bounds__(256) void convert_f32_to_f16(const float* __restrict__ src,
                                                          half_t* __restrict__ dst, int n) {
  int i = (blockIdx.x * blockDim.x + threadIdx.x) * 4;
  if (i + 3 < n) {
    float4 v = *(const float4*)(src + i);
    union { half_t h[4]; uint2 q; } u;
    u.h[0] = (half_t)v.x; u.h[1] = (half_t)v.y; u.h[2] = (half_t)v.z; u.h[3] = (half_t)v.w;
    *(uint2*)(dst + i) = u.q;
  }
}

// ---------------------------------------------------------------------------
// Core 128x128 tile GEMM:  C[m,n] = sum_k A[m,k]*B[n,k] + bias[n]
// A rows [row0, rowEnd) (clamped/masked), B row n = weight row (y = x @ W^T).
// 8 waves: wave (wm, wn) owns a 32x64 sub-tile = 2x4 WMMA accumulators.
// Double-buffered K panels filled by async global->LDS copies.
// ---------------------------------------------------------------------------
template <bool RELU_HALF_OUT>
__device__ __forceinline__ void gemm_tile(const half_t* __restrict__ A,
                                          int row0, int rowEnd,
                                          const half_t* __restrict__ B, int nb0,
                                          const float* __restrict__ bias,
                                          half_t* __restrict__ outH,
                                          float* __restrict__ outF, int ldc) {
  __shared__ half_t As[2][TM][LDSW];
  __shared__ half_t Bs[2][TN][LDSW];

  const int tid  = threadIdx.x;
  const int lane = tid & 31;
  const int wave = tid >> 5;       // 0..7
  const int wm   = wave & 3;       // 4 waves along M
  const int wn   = wave >> 2;      // 2 waves along N
  const int hsel = lane >> 4;      // half-select (K phase)
  const int r    = lane & 15;      // row-in-tile / col-in-tile

  // global->LDS copy assignment: 2 threads per row, 8 halfs (16B) each, at half offset 0 or 16
  const int crow = tid >> 1;
  const int coff = (tid & 1) * 16;
  int arow = row0 + crow;
  const int maxrow = rowEnd - 1;
  if (arow > maxrow) arow = maxrow;            // clamp padded rows (masked on store)
  const half_t* aSrc = A + (size_t)arow * DDIM + coff;
  const half_t* bSrc = B + (size_t)(nb0 + crow) * DDIM + coff;

  v8f acc[2][4] = {};

  // prologue: panel 0 (async -> LDS, drain, publish)
  async_copy_b128(aSrc, &As[0][crow][coff]);
  async_copy_b128(bSrc, &Bs[0][crow][coff]);
  wait_async0();
  __syncthreads();

#pragma unroll 1
  for (int kt = 0; kt < KSTEPS; ++kt) {
    const int cur = kt & 1;
    if (kt + 1 < KSTEPS) {
      // issue next panel early; it lands under the WMMA work below
      async_copy_b128(aSrc + (size_t)(kt + 1) * TK, &As[cur ^ 1][crow][coff]);
      async_copy_b128(bSrc + (size_t)(kt + 1) * TK, &Bs[cur ^ 1][crow][coff]);
    }

    v16h a[2], b[4];
#pragma unroll
    for (int mt = 0; mt < 2; ++mt) {
      const half_t* p = &As[cur][wm * 32 + mt * 16 + r][hsel * 8];
      union { v16h v; uint4 q[2]; } u;
      u.q[0] = *(const uint4*)p;
      u.q[1] = *(const uint4*)(p + 16);
      a[mt] = u.v;
    }
#pragma unroll
    for (int nt = 0; nt < 4; ++nt) {
      const half_t* p = &Bs[cur][wn * 64 + nt * 16 + r][hsel * 8];
      union { v16h v; uint4 q[2]; } u;
      u.q[0] = *(const uint4*)p;
      u.q[1] = *(const uint4*)(p + 16);
      b[nt] = u.v;
    }
#pragma unroll
    for (int mt = 0; mt < 2; ++mt)
#pragma unroll
      for (int nt = 0; nt < 4; ++nt)
        acc[mt][nt] = __builtin_amdgcn_wmma_f32_16x16x32_f16(
            false, a[mt], false, b[nt], (short)0, acc[mt][nt], false, false);

    wait_async0();     // our next-panel copies have landed in LDS
    __syncthreads();   // everyone's copies visible; everyone done reading cur
  }

  // epilogue: D layout: VGPR e -> M = hsel*8 + e, N = r (per 16x16 tile)
#pragma unroll
  for (int nt = 0; nt < 4; ++nt) {
    const int n = nb0 + wn * 64 + nt * 16 + r;
    const float bn = bias[n];
#pragma unroll
    for (int mt = 0; mt < 2; ++mt) {
#pragma unroll
      for (int e = 0; e < 8; ++e) {
        const int m = row0 + wm * 32 + mt * 16 + hsel * 8 + e;
        if (m < rowEnd) {
          float v = acc[mt][nt][e] + bn;
          if constexpr (RELU_HALF_OUT) {
            v = v > 0.0f ? v : 0.0f;
            outH[(size_t)m * ldc + n] = (half_t)v;
          } else {
            outF[(size_t)m * ldc + n] = v;
          }
        }
      }
    }
  }
}

// GEMM1: per-scene  ys[p,:] = relu(xs[p,:] @ Ws[s]^T + bs[s])  (fp16 out)
__global__ __launch_bounds__(256) void gemm1_scene(const half_t* __restrict__ xs,
                                                   const half_t* __restrict__ Wsh,
                                                   const float* __restrict__ bs,
                                                   const int* __restrict__ meta,
                                                   half_t* __restrict__ ys) {
  const int t = blockIdx.y;
  int g[5], ts[4];
#pragma unroll
  for (int i = 0; i < 5; ++i) g[i] = meta[i];
#pragma unroll
  for (int s = 0; s < 4; ++s) ts[s] = meta[5 + s];
  if (t >= meta[9]) return;
  int s = 0;
#pragma unroll
  for (int i = 1; i < 4; ++i)
    if (t >= ts[i]) s = i;
  const int row0 = g[s] + (t - ts[s]) * TM;
  const int rowEnd = g[s + 1];
  gemm_tile<true>(xs, row0, rowEnd,
                  Wsh + (size_t)s * DDIM * DDIM, blockIdx.x * TN,
                  bs + (size_t)s * DDIM, ys, nullptr, DDIM);
}

// GEMM2: dense  out = ys @ W1^T + b1  (fp32 out, rows already in sorted order)
__global__ __launch_bounds__(256) void gemm2_dense(const half_t* __restrict__ ys,
                                                   const half_t* __restrict__ W1h,
                                                   const float* __restrict__ b1,
                                                   float* __restrict__ out) {
  gemm_tile<false>(ys, blockIdx.y * TM, N_ROWS,
                   W1h, blockIdx.x * TN, b1, nullptr, out, HDIM);
}

extern "C" void kernel_launch(void* const* d_in, const int* in_sizes, int n_in,
                              void* d_out, int out_size, void* d_ws, size_t ws_size,
                              hipStream_t stream) {
  const float* x     = (const float*)d_in[0];
  const int*   scene = (const int*)d_in[1];
  const float* Ws    = (const float*)d_in[2];
  const float* bs    = (const float*)d_in[3];
  const float* W1    = (const float*)d_in[4];
  const float* b1    = (const float*)d_in[5];
  float* out = (float*)d_out;

  // workspace carve-up (~561 MB)
  char* ws = (char*)d_ws;
  int* sorted_idx = (int*)ws;                                   // N ints
  int* meta = (int*)(ws + (size_t)N_ROWS * 4);                  // 16 ints
  size_t off = (((size_t)N_ROWS * 4 + 64) + 255) & ~(size_t)255;
  half_t* xs  = (half_t*)(ws + off); off += (size_t)N_ROWS * DDIM * 2;
  half_t* Wsh = (half_t*)(ws + off); off += (size_t)NSCENE * DDIM * DDIM * 2;
  half_t* W1h = (half_t*)(ws + off); off += (size_t)HDIM * DDIM * 2;
  half_t* ysh = (half_t*)(ws + off); off += (size_t)N_ROWS * DDIM * 2;

  sort_scenes<<<1, 256, 0, stream>>>(scene, sorted_idx, meta);
  gather_convert_x<<<N_ROWS, 256, 0, stream>>>(x, sorted_idx, xs);
  {
    int n = NSCENE * DDIM * DDIM;
    convert_f32_to_f16<<<n / 1024, 256, 0, stream>>>(Ws, Wsh, n);
    n = HDIM * DDIM;
    convert_f32_to_f16<<<n / 1024, 256, 0, stream>>>(W1, W1h, n);
  }
  dim3 g1(DDIM / TN, N_ROWS / TM + NSCENE);   // +NSCENE covers partial-tile padding
  gemm1_scene<<<g1, 256, 0, stream>>>(xs, Wsh, bs, meta, ysh);
  dim3 g2(HDIM / TN, N_ROWS / TM);
  gemm2_dense<<<g2, 256, 0, stream>>>(ysh, W1h, b1, out);
}
